// MixtureOfExperts_62096637165902
// MI455X (gfx1250) — matrix operand order repaired and checked by reading, not compiled, not run on hardware
//
#include <hip/hip_runtime.h>

#define DIN  1024
#define HID  2048
#define OUTD 1024
#define NEXP 8
#define NTOK 8192

typedef _Float16 v16h __attribute__((ext_vector_type(16)));
typedef _Float16 v8h  __attribute__((ext_vector_type(8)));
typedef float    v8f  __attribute__((ext_vector_type(8)));

constexpr int SA = 72;   // LDS row stride (halves): 64 K + pad, 144B rows (16B aligned)
constexpr int SB = 72;   // LDS row stride (halves) for B tile (stored [n][k])

// ---------------------------------------------------------------- init
__global__ __launch_bounds__(64) void moe_init(int* __restrict__ cnt) {
  if (threadIdx.x < NEXP) cnt[threadIdx.x] = 0;
}

// ---------------------------------------------------------------- router
// One wave (32 lanes) per token: 8 dot products over D=1024, wave reduce,
// exact fp32 softmax, first-occurrence argmax, atomic scatter to expert list.
__global__ __launch_bounds__(256) void moe_router(
    const float* __restrict__ x, const float* __restrict__ Wr,
    const float* __restrict__ br, float* __restrict__ probs,
    int* __restrict__ tokList, int* __restrict__ cnt) {
  const int lane  = threadIdx.x & 31;
  const int wave  = threadIdx.x >> 5;
  const int token = blockIdx.x * 8 + wave;
  const float* xr = x + (size_t)token * DIN;

  float acc[NEXP];
#pragma unroll
  for (int e = 0; e < NEXP; ++e) acc[e] = 0.f;

  for (int i = lane; i < DIN; i += 32) {
    const float xv  = xr[i];
    const float4 w0 = *(const float4*)(Wr + (size_t)i * NEXP);
    const float4 w1 = *(const float4*)(Wr + (size_t)i * NEXP + 4);
    acc[0] += xv * w0.x; acc[1] += xv * w0.y;
    acc[2] += xv * w0.z; acc[3] += xv * w0.w;
    acc[4] += xv * w1.x; acc[5] += xv * w1.y;
    acc[6] += xv * w1.z; acc[7] += xv * w1.w;
  }
#pragma unroll
  for (int off = 16; off > 0; off >>= 1) {
#pragma unroll
    for (int e = 0; e < NEXP; ++e) acc[e] += __shfl_xor(acc[e], off, 32);
  }

  if (lane == 0) {
    float z[NEXP];
#pragma unroll
    for (int e = 0; e < NEXP; ++e) z[e] = acc[e] + br[e];
    float m = z[0]; int best = 0;
#pragma unroll
    for (int e = 1; e < NEXP; ++e) if (z[e] > m) { m = z[e]; best = e; }
    float p[NEXP]; float s = 0.f;
#pragma unroll
    for (int e = 0; e < NEXP; ++e) { p[e] = expf(z[e] - m); s += p[e]; }
    const float inv = 1.0f / s;
#pragma unroll
    for (int e = 0; e < NEXP; ++e) probs[(size_t)token * NEXP + e] = p[e] * inv;
    const int pos = atomicAdd(&cnt[best], 1);
    tokList[best * NTOK + pos] = token;
  }
}

// ---------------------------------------------------------------- counts
__global__ __launch_bounds__(64) void moe_counts(const int* __restrict__ cnt,
                                                 float* __restrict__ outCounts) {
  if (threadIdx.x < NEXP) outCounts[threadIdx.x] = (float)cnt[threadIdx.x];
}

// ================================================================
// Shared GEMM geometry: 256 threads = 8 waves arranged 4(m) x 2(n).
// Wave tile 32x64 = 2x4 subtiles of v_wmma_f32_16x16x32_f16.
// Block tile 128(M) x 128(N), K staged 64 at a time through LDS.
// Fragment layouts per ISA 7.12.2:
//   A lane(l): m=l&15, halves j=0..7 -> K=hi*8+j, j=8..15 -> K=16+hi*8+j
//   B lane(l): n=l&15, halves j=0..7 -> K=hi*8+j, j=8..15 -> K=16+hi*8+j  (so LDS B stored [n][k])
//   C/D lane(l), elem r: M = r + 8*hi, N = l&15
// ================================================================

// ---------------------------------------------------------------- GEMM1
// h[tok, n] = relu( x[tok, :] @ W1[e] + b1[e] ), fp16 out, gathered rows.
__global__ __launch_bounds__(256) void moe_gemm1(
    const float* __restrict__ x, const float* __restrict__ W1,
    const float* __restrict__ b1, const int* __restrict__ tokList,
    const int* __restrict__ cnt, _Float16* __restrict__ hbuf) {
  const int e    = blockIdx.z;
  const int cntE = cnt[e];
  const int m0   = blockIdx.y * 128;
  if (m0 >= cntE) return;
  const int n0   = blockIdx.x * 128;

  __shared__ _Float16 As[128 * SA];
  __shared__ _Float16 Bs[128 * SB];
  __shared__ int sTok[128];

  const int tid    = threadIdx.x;
  const int lane   = tid & 31;
  const int wave   = tid >> 5;
  const int lane15 = lane & 15;
  const int hi     = lane >> 4;
  const int wm0    = (wave & 3) * 32;   // 4 m-waves
  const int wn0    = (wave >> 2) * 64;  // 2 n-waves

  if (tid < 128) {
    const int row = m0 + tid;
    sTok[tid] = (row < cntE) ? tokList[e * NTOK + row] : -1;
  }
  __syncthreads();

  const float* __restrict__ W1e = W1 + (size_t)e * DIN * HID;
  v8f acc[2][4] = {};

  for (int k0 = 0; k0 < DIN; k0 += 64) {
    // A tile: 128 rows x 64 K, fp32 gather -> fp16 LDS (2048 float4)
#pragma unroll
    for (int it = 0; it < 8; ++it) {
      const int i = tid + it * 256;
      const int r = i >> 4;
      const int c = (i & 15) * 4;
      int t = sTok[r]; t = (t < 0) ? 0 : t;
      const float4 v = *(const float4*)(x + (size_t)t * DIN + k0 + c);
      _Float16* dst = &As[r * SA + c];
      dst[0] = (_Float16)v.x; dst[1] = (_Float16)v.y;
      dst[2] = (_Float16)v.z; dst[3] = (_Float16)v.w;
    }
    // B tile transposed into LDS: Bs[n][k] <- W1e[k0+kr][n0+c] (2048 float4)
#pragma unroll
    for (int it = 0; it < 8; ++it) {
      const int i  = tid + it * 256;
      const int kr = i >> 5;
      const int c  = (i & 31) * 4;
      const float4 v = *(const float4*)(W1e + (size_t)(k0 + kr) * HID + n0 + c);
      Bs[(c + 0) * SB + kr] = (_Float16)v.x;
      Bs[(c + 1) * SB + kr] = (_Float16)v.y;
      Bs[(c + 2) * SB + kr] = (_Float16)v.z;
      Bs[(c + 3) * SB + kr] = (_Float16)v.w;
    }
    __syncthreads();

#pragma unroll
    for (int kk = 0; kk < 64; kk += 32) {
      v16h a[2], b[4];
#pragma unroll
      for (int mi = 0; mi < 2; ++mi) {
        const _Float16* ap = &As[(wm0 + mi * 16 + lane15) * SA + kk + hi * 8];
        v8h lo = *(const v8h*)ap;
        v8h hv = *(const v8h*)(ap + 16);
        a[mi] = __builtin_shufflevector(lo, hv, 0,1,2,3,4,5,6,7,8,9,10,11,12,13,14,15);
      }
#pragma unroll
      for (int ni = 0; ni < 4; ++ni) {
        const _Float16* bp = &Bs[(wn0 + ni * 16 + lane15) * SB + kk + hi * 8];
        v8h lo = *(const v8h*)bp;
        v8h hv = *(const v8h*)(bp + 16);
        b[ni] = __builtin_shufflevector(lo, hv, 0,1,2,3,4,5,6,7,8,9,10,11,12,13,14,15);
      }
#pragma unroll
      for (int mi = 0; mi < 2; ++mi)
#pragma unroll
        for (int ni = 0; ni < 4; ++ni)
          acc[mi][ni] = __builtin_amdgcn_wmma_f32_16x16x32_f16(
              false, a[mi], false, b[ni], (short)0, acc[mi][ni], false, false);
    }
    __syncthreads();
  }

  const float* __restrict__ b1e = b1 + (size_t)e * HID;
#pragma unroll
  for (int mi = 0; mi < 2; ++mi) {
#pragma unroll
    for (int ni = 0; ni < 4; ++ni) {
      const int col  = n0 + wn0 + ni * 16 + lane15;
      const float bv = b1e[col];
#pragma unroll
      for (int r = 0; r < 8; ++r) {
        const int rowL = wm0 + mi * 16 + r + hi * 8;
        const int t = sTok[rowL];
        if (t >= 0) {
          float v = acc[mi][ni][r] + bv;
          v = v > 0.f ? v : 0.f;
          hbuf[(size_t)t * HID + col] = (_Float16)v;
        }
      }
    }
  }
}

// ---------------------------------------------------------------- GEMM2
// out[tok, n] = h[tok, :] @ W2[e] + b2[e], fp32 output to d_out.
__global__ __launch_bounds__(256) void moe_gemm2(
    const _Float16* __restrict__ hbuf, const float* __restrict__ W2,
    const float* __restrict__ b2, const int* __restrict__ tokList,
    const int* __restrict__ cnt, float* __restrict__ out) {
  const int e    = blockIdx.z;
  const int cntE = cnt[e];
  const int m0   = blockIdx.y * 128;
  if (m0 >= cntE) return;
  const int n0   = blockIdx.x * 128;

  __shared__ _Float16 As[128 * SA];
  __shared__ _Float16 Bs[128 * SB];
  __shared__ int sTok[128];

  const int tid    = threadIdx.x;
  const int lane   = tid & 31;
  const int wave   = tid >> 5;
  const int lane15 = lane & 15;
  const int hi     = lane >> 4;
  const int wm0    = (wave & 3) * 32;
  const int wn0    = (wave >> 2) * 64;

  if (tid < 128) {
    const int row = m0 + tid;
    sTok[tid] = (row < cntE) ? tokList[e * NTOK + row] : -1;
  }
  __syncthreads();

  const float* __restrict__ W2e = W2 + (size_t)e * HID * OUTD;
  v8f acc[2][4] = {};

  for (int k0 = 0; k0 < HID; k0 += 64) {
    // A tile: 128 rows x 64 K fp16 gather (8B = 4 halves per copy)
#pragma unroll
    for (int it = 0; it < 8; ++it) {
      const int i = tid + it * 256;
      const int r = i >> 4;
      const int c = (i & 15) * 4;
      int t = sTok[r]; t = (t < 0) ? 0 : t;
      const uint2 v = *(const uint2*)(hbuf + (size_t)t * HID + k0 + c);
      *(uint2*)&As[r * SA + c] = v;
    }
    // B tile transposed: Bs[n][k] <- W2e[k0+kr][n0+c]
#pragma unroll
    for (int it = 0; it < 8; ++it) {
      const int i  = tid + it * 256;
      const int kr = i >> 5;
      const int c  = (i & 31) * 4;
      const float4 v = *(const float4*)(W2e + (size_t)(k0 + kr) * OUTD + n0 + c);
      Bs[(c + 0) * SB + kr] = (_Float16)v.x;
      Bs[(c + 1) * SB + kr] = (_Float16)v.y;
      Bs[(c + 2) * SB + kr] = (_Float16)v.z;
      Bs[(c + 3) * SB + kr] = (_Float16)v.w;
    }
    __syncthreads();

#pragma unroll
    for (int kk = 0; kk < 64; kk += 32) {
      v16h a[2], b[4];
#pragma unroll
      for (int mi = 0; mi < 2; ++mi) {
        const _Float16* ap = &As[(wm0 + mi * 16 + lane15) * SA + kk + hi * 8];
        v8h lo = *(const v8h*)ap;
        v8h hv = *(const v8h*)(ap + 16);
        a[mi] = __builtin_shufflevector(lo, hv, 0,1,2,3,4,5,6,7,8,9,10,11,12,13,14,15);
      }
#pragma unroll
      for (int ni = 0; ni < 4; ++ni) {
        const _Float16* bp = &Bs[(wn0 + ni * 16 + lane15) * SB + kk + hi * 8];
        v8h lo = *(const v8h*)bp;
        v8h hv = *(const v8h*)(bp + 16);
        b[ni] = __builtin_shufflevector(lo, hv, 0,1,2,3,4,5,6,7,8,9,10,11,12,13,14,15);
      }
#pragma unroll
      for (int mi = 0; mi < 2; ++mi)
#pragma unroll
        for (int ni = 0; ni < 4; ++ni)
          acc[mi][ni] = __builtin_amdgcn_wmma_f32_16x16x32_f16(
              false, a[mi], false, b[ni], (short)0, acc[mi][ni], false, false);
    }
    __syncthreads();
  }

  const float* __restrict__ b2e = b2 + (size_t)e * OUTD;
#pragma unroll
  for (int mi = 0; mi < 2; ++mi) {
#pragma unroll
    for (int ni = 0; ni < 4; ++ni) {
      const int col  = n0 + wn0 + ni * 16 + lane15;
      const float bv = b2e[col];
#pragma unroll
      for (int r = 0; r < 8; ++r) {
        const int rowL = wm0 + mi * 16 + r + hi * 8;
        const int t = sTok[rowL];
        if (t >= 0) out[(size_t)t * OUTD + col] = acc[mi][ni][r] + bv;
      }
    }
  }
}

// ---------------------------------------------------------------- launch
extern "C" void kernel_launch(void* const* d_in, const int* in_sizes, int n_in,
                              void* d_out, int out_size, void* d_ws, size_t ws_size,
                              hipStream_t stream) {
  (void)in_sizes; (void)n_in; (void)out_size; (void)ws_size;
  const float* x  = (const float*)d_in[0];
  const float* Wr = (const float*)d_in[1];
  const float* br = (const float*)d_in[2];
  const float* W1 = (const float*)d_in[3];
  const float* b1 = (const float*)d_in[4];
  const float* W2 = (const float*)d_in[5];
  const float* b2 = (const float*)d_in[6];

  float* out       = (float*)d_out;
  float* probs     = out + (size_t)NTOK * OUTD;
  float* countsOut = probs + (size_t)NTOK * NEXP;

  int* cnt        = (int*)d_ws;
  int* tokList    = cnt + NEXP;
  _Float16* hbuf  = (_Float16*)(tokList + NEXP * NTOK);  // NTOK*HID fp16 = 32 MB

  moe_init  <<<1, 64, 0, stream>>>(cnt);
  moe_router<<<NTOK / 8, 256, 0, stream>>>(x, Wr, br, probs, tokList, cnt);
  moe_counts<<<1, 64, 0, stream>>>(cnt, countsOut);
  moe_gemm1 <<<dim3(HID / 128,  NTOK / 128, NEXP), 256, 0, stream>>>(x, W1, b1, tokList, cnt, hbuf);
  moe_gemm2 <<<dim3(OUTD / 128, NTOK / 128, NEXP), 256, 0, stream>>>(hbuf, W2, b2, tokList, cnt, out);
}
